// MultiHeadAttention_27032524161249
// MI455X (gfx1250) — matrix operand order repaired and checked by reading, not compiled, not run on hardware
//
#include <hip/hip_runtime.h>
#include <hip/hip_bf16.h>

// ---------------------------------------------------------------------------
// MHA forward for gfx1250 (CDNA5, wave32, WMMA).
// B=16, S=1024, D=256, H=8, DK=DV=64.
// All GEMM B-operands are pre-transposed so every B-fragment is a contiguous
// 32-byte load per lane (contiguous k = 16*half + s, same order as the A loads).
// ---------------------------------------------------------------------------

typedef __bf16 bf16_t;
typedef __attribute__((ext_vector_type(16))) __bf16 v16bf;
typedef __attribute__((ext_vector_type(8)))  __bf16 v8bf;
typedef __attribute__((ext_vector_type(8)))  float  v8f;

#define WMMA_BF16(a, b, c) \
    __builtin_amdgcn_wmma_f32_16x16x32_bf16(false, (a), false, (b), (short)0, (c), false, false)

constexpr int BB = 16;
constexpr int SS = 1024;
constexpr int DD = 256;
constexpr int HH = 8;
constexpr int DKV = 64;          // DK == DV
constexpr int HD = HH * DKV;     // 512
constexpr float NEGV = -1e9f;
constexpr float EPS = 1e-5f;

// ---------------------------------------------------------------------------
// Fragment loaders (A/B in this file's contiguous-k order, wave32)
// A 16x32 bf16: lane L(0-15) row=L, K=0..15 contiguous; lane L+16 row=L, K=16..31
// B 32x16 bf16: lanes 0-15 col=L hold K=0..15 contiguous; lanes 16-31 hold K=16..31
// C/D 16x16 f32: vgpr r: lanes 0-15 -> M=r, lanes 16-31 -> M=8+r; N=lane&15
// ---------------------------------------------------------------------------

__device__ __forceinline__ v16bf load_a_frag(const bf16_t* __restrict__ base,
                                             int m0, int k0, int ld, int lane) {
    const int row  = m0 + (lane & 15);
    const int half = lane >> 4;
    const bf16_t* p = base + (size_t)row * ld + k0 + 16 * half;
    v8bf lo = *reinterpret_cast<const v8bf*>(p);
    v8bf hi = *reinterpret_cast<const v8bf*>(p + 8);
    v16bf a;
#pragma unroll
    for (int i = 0; i < 8; ++i) { a[i] = lo[i]; a[i + 8] = hi[i]; }
    return a;
}

// B operand where columns of B are contiguous rows in memory (memory holds B^T
// row-major with row stride ld): one contiguous 32-byte load per lane.
__device__ __forceinline__ v16bf load_b_frag_rows(const bf16_t* __restrict__ base,
                                                  int n0, int k0, int ld, int lane) {
    const int col  = n0 + (lane & 15);
    const int half = lane >> 4;
    const bf16_t* p = base + (size_t)col * ld + k0 + 16 * half;
    return *reinterpret_cast<const v16bf*>(p);
}

// A operand from bf16 LDS (attention probs): two 16-byte ds loads per lane.
__device__ __forceinline__ v16bf load_a_frag_lds(const bf16_t* sm, int k0, int lane) {
    const int row  = lane & 15;
    const int half = lane >> 4;
    const bf16_t* p = sm + row * SS + k0 + 16 * half;
    v8bf lo = *reinterpret_cast<const v8bf*>(p);
    v8bf hi = *reinterpret_cast<const v8bf*>(p + 8);
    v16bf a;
#pragma unroll
    for (int i = 0; i < 8; ++i) { a[i] = lo[i]; a[i + 8] = hi[i]; }
    return a;
}

// ---------------------------------------------------------------------------
// Kernel 0a: fp32 -> bf16 (elementwise)
// ---------------------------------------------------------------------------
__global__ void f32_to_bf16_kernel(const float* __restrict__ in,
                                   bf16_t* __restrict__ out, int n) {
    int i = blockIdx.x * blockDim.x + threadIdx.x;
    if (i < n) out[i] = (bf16_t)in[i];
}

// ---------------------------------------------------------------------------
// Kernel 0b: fp32 [K][N] -> bf16 [N][K] (transpose + convert)
// ---------------------------------------------------------------------------
__global__ void f32_to_bf16_T_kernel(const float* __restrict__ in,
                                     bf16_t* __restrict__ out, int K, int N) {
    int idx = blockIdx.x * blockDim.x + threadIdx.x;
    if (idx < K * N) {
        int k = idx / N;
        int n = idx - k * N;
        out[(size_t)n * K + k] = (bf16_t)in[idx];
    }
}

// ---------------------------------------------------------------------------
// Kernel 1: projection GEMM  P[16384,512] = X[16384,256] @ W[256,512]
// WT = W^T bf16 [512][256].  One wave per 16x64 strip (4 acc tiles, A reused).
// mode 0/1: write [B,H,S,64] (Q, K);  mode 2: write [B,H,64,S] (V transposed)
// ---------------------------------------------------------------------------
__global__ __launch_bounds__(256) void proj_gemm_kernel(const bf16_t* __restrict__ X,
                                                        const bf16_t* __restrict__ WT,
                                                        bf16_t* __restrict__ out,
                                                        int mode) {
    const int tid  = threadIdx.x;
    const int lane = tid & 31;
    const int half = lane >> 4;
    const int wave = tid >> 5;
    const int strip = blockIdx.x * 8 + wave;    // 8192 strips (1024 x 8)
    const int mt = strip >> 3;
    const int nt = strip & 7;
    const int m0 = mt * 16;
    const int n0 = nt * 64;

    v8f acc[4] = {};
#pragma unroll
    for (int k0 = 0; k0 < DD; k0 += 32) {
        v16bf a = load_a_frag(X, m0, k0, DD, lane);
#pragma unroll
        for (int j = 0; j < 4; ++j) {
            v16bf b = load_b_frag_rows(WT, n0 + 16 * j, k0, DD, lane);
            acc[j] = WMMA_BF16(a, b, acc[j]);
        }
    }

#pragma unroll
    for (int j = 0; j < 4; ++j) {
#pragma unroll
        for (int r = 0; r < 8; ++r) {
            const int row = m0 + r + 8 * half;       // M in [0,16384)
            const int col = n0 + 16 * j + (lane & 15); // N in [0,512)
            const int bi = row >> 10;
            const int s  = row & (SS - 1);
            const int h  = col >> 6;
            const int d  = col & (DKV - 1);
            bf16_t v = (bf16_t)acc[j][r];
            if (mode == 2) {
                out[(((size_t)bi * HH + h) * DKV + d) * SS + s] = v;  // [B,H,DV,S]
            } else {
                out[(((size_t)bi * HH + h) * SS + s) * DKV + d] = v;  // [B,H,S,DK]
            }
        }
    }
}

// ---------------------------------------------------------------------------
// Kernel 2: attention.  One block (128 thr, 4 waves) per (b, h, 16-query tile).
// scores -> LDS f32 (64KB) -> parallel softmax -> attn to global + bf16 LDS
// probs (32KB) -> attn @ V via WMMA -> prob bf16 [B*S, H*DV]
// ---------------------------------------------------------------------------
__global__ __launch_bounds__(128) void attn_kernel(const bf16_t* __restrict__ Qh,
                                                   const bf16_t* __restrict__ Kh,
                                                   const bf16_t* __restrict__ VT,
                                                   const unsigned char* __restrict__ mask,
                                                   float* __restrict__ attn_out,
                                                   bf16_t* __restrict__ prob) {
    __shared__ float  sc[16 * SS];     // 64 KB fp32 scores / probs
    __shared__ bf16_t scb[16 * SS];    // 32 KB bf16 probs for WMMA A operand
    __shared__ float  red[16 * 8];     // partial reductions
    __shared__ float  rowv[16];        // row max / inv-sum

    const int tid  = threadIdx.x;
    const int lane = tid & 31;
    const int half = lane >> 4;
    const int wave = tid >> 5;

    const int qt = blockIdx.x & 63;
    const int h  = (blockIdx.x >> 6) & 7;
    const int b  = blockIdx.x >> 9;
    const int q0 = qt * 16;

    const bf16_t* Qb = Qh + (((size_t)b * HH + h) * SS) * DKV;   // [S][64]
    const bf16_t* Kb = Kh + (((size_t)b * HH + h) * SS) * DKV;   // [S][64]
    const bf16_t* Vb = VT + (((size_t)b * HH + h) * DKV) * SS;   // [64][S]

    // Q fragments (K=64 -> two K=32 steps), reused across all 64 key tiles.
    v16bf aq0 = load_a_frag(Qb, q0, 0, DKV, lane);
    v16bf aq1 = load_a_frag(Qb, q0, 32, DKV, lane);

    // ---- scores = (Q K^T) / sqrt(64); each wave covers 16 key tiles ----
    for (int ntile = wave; ntile < 64; ntile += 4) {
        const int n0 = ntile * 16;
        __builtin_prefetch(Kb + (size_t)(n0 + 64) * DKV, 0, 1);
        v8f acc = {};
        v16bf b0 = load_b_frag_rows(Kb, n0, 0, DKV, lane);
        v16bf b1 = load_b_frag_rows(Kb, n0, 32, DKV, lane);
        acc = WMMA_BF16(aq0, b0, acc);
        acc = WMMA_BF16(aq1, b1, acc);
#pragma unroll
        for (int r = 0; r < 8; ++r) {
            const int rq  = r + 8 * half;
            const int key = n0 + (lane & 15);
            float v = acc[r] * 0.125f;           // 1/sqrt(DK)
            if (mask[((size_t)b * SS + (q0 + rq)) * SS + key]) v = NEGV;
            sc[rq * SS + key] = v;
        }
    }
    __syncthreads();

    // ---- softmax: 8 threads per row, 128-element chunks ----
    const int srow  = tid & 15;
    const int chunk = tid >> 4;                   // 0..7
    float* rp = &sc[srow * SS + chunk * 128];
    {
        float m = rp[0];
        for (int i = 1; i < 128; ++i) m = fmaxf(m, rp[i]);
        red[srow * 8 + chunk] = m;
    }
    __syncthreads();
    if (tid < 16) {
        float m = red[tid * 8];
        for (int j = 1; j < 8; ++j) m = fmaxf(m, red[tid * 8 + j]);
        rowv[tid] = m;
    }
    __syncthreads();
    {
        const float rm = rowv[srow];
        float s = 0.0f;
        for (int i = 0; i < 128; ++i) {
            float e = __expf(rp[i] - rm);
            rp[i] = e;
            s += e;
        }
        red[srow * 8 + chunk] = s;
    }
    __syncthreads();
    if (tid < 16) {
        float s = 0.0f;
        for (int j = 0; j < 8; ++j) s += red[tid * 8 + j];
        rowv[tid] = 1.0f / s;
    }
    __syncthreads();

    // ---- normalize + write attn [B,H,S,S] + stage bf16 probs in LDS ----
    {
        const float inv = rowv[srow];
        float* aout = attn_out + (((size_t)b * HH + h) * SS + q0 + srow) * SS + chunk * 128;
        bf16_t* sb  = &scb[srow * SS + chunk * 128];
        for (int i = 0; i < 128; ++i) {
            float v = rp[i] * inv;
            aout[i] = v;
            sb[i] = (bf16_t)v;
        }
    }
    __syncthreads();

    // ---- prob = attn @ V : each wave computes one 16x16 dv tile ----
    const int n0 = wave * 16;
    v8f acc = {};
    for (int kt = 0; kt < SS / 32; ++kt) {
        const int k0 = kt * 32;
        __builtin_prefetch(Vb + (size_t)(n0 + (lane & 15)) * SS + k0 + 128, 0, 1);
        v16bf a  = load_a_frag_lds(scb, k0, lane);
        v16bf bv = load_b_frag_rows(Vb, n0, k0, SS, lane);
        acc = WMMA_BF16(a, bv, acc);
    }
#pragma unroll
    for (int r = 0; r < 8; ++r) {
        const int rq = r + 8 * half;
        const int dv = n0 + (lane & 15);
        // prob layout: [B*S, H*DV] row-major -> feeds output projection GEMM
        prob[((size_t)b * SS + q0 + rq) * HD + h * DKV + dv] = (bf16_t)acc[r];
    }
}

// ---------------------------------------------------------------------------
// Kernel 3: out = LayerNorm(input_Q + prob @ W_O)
// Block = 128 thr (4 waves) per 16 rows; each wave computes a 16x64 strip.
// WoT = W_O^T bf16 [256][512].
// ---------------------------------------------------------------------------
__global__ __launch_bounds__(128) void out_ln_kernel(const bf16_t* __restrict__ P,
                                                     const bf16_t* __restrict__ WoT,
                                                     const float* __restrict__ Xres,
                                                     const float* __restrict__ gamma,
                                                     const float* __restrict__ beta,
                                                     float* __restrict__ out) {
    __shared__ float xb[16 * DD];               // 16 KB
    __shared__ float red[16 * 8];
    __shared__ float rowm[16], rowr[16];

    const int tid  = threadIdx.x;
    const int lane = tid & 31;
    const int half = lane >> 4;
    const int wave = tid >> 5;
    const int m0   = blockIdx.x * 16;
    const int n0   = wave * 64;

    v8f acc[4] = {};
    for (int k0 = 0; k0 < HD; k0 += 32) {
        v16bf a = load_a_frag(P, m0, k0, HD, lane);
#pragma unroll
        for (int j = 0; j < 4; ++j) {
            v16bf b = load_b_frag_rows(WoT, n0 + 16 * j, k0, HD, lane);
            acc[j] = WMMA_BF16(a, b, acc[j]);
        }
    }
#pragma unroll
    for (int j = 0; j < 4; ++j) {
#pragma unroll
        for (int r = 0; r < 8; ++r) {
            const int rl  = r + 8 * half;
            const int col = n0 + 16 * j + (lane & 15);
            xb[rl * DD + col] = acc[j][r] + Xres[(size_t)(m0 + rl) * DD + col];
        }
    }
    __syncthreads();

    // ---- LayerNorm: 8 threads per row, 32-element chunks ----
    const int srow  = tid & 15;
    const int chunk = tid >> 4;
    const float* rp = &xb[srow * DD + chunk * 32];
    {
        float s = 0.0f;
        for (int i = 0; i < 32; ++i) s += rp[i];
        red[srow * 8 + chunk] = s;
    }
    __syncthreads();
    if (tid < 16) {
        float s = 0.0f;
        for (int j = 0; j < 8; ++j) s += red[tid * 8 + j];
        rowm[tid] = s * (1.0f / DD);
    }
    __syncthreads();
    {
        const float mu = rowm[srow];
        float s = 0.0f;
        for (int i = 0; i < 32; ++i) {
            const float d = rp[i] - mu;
            s += d * d;
        }
        red[srow * 8 + chunk] = s;
    }
    __syncthreads();
    if (tid < 16) {
        float s = 0.0f;
        for (int j = 0; j < 8; ++j) s += red[tid * 8 + j];
        rowr[tid] = rsqrtf(s * (1.0f / DD) + EPS);
    }
    __syncthreads();
    {
        const float mu = rowm[srow];
        const float rstd = rowr[srow];
        float* o = out + (size_t)(m0 + srow) * DD + chunk * 32;
        const float* g = gamma + chunk * 32;
        const float* be = beta + chunk * 32;
        for (int i = 0; i < 32; ++i)
            o[i] = (rp[i] - mu) * rstd * g[i] + be[i];
    }
}

// ---------------------------------------------------------------------------
// Host launcher
// ---------------------------------------------------------------------------
extern "C" void kernel_launch(void* const* d_in, const int* in_sizes, int n_in,
                              void* d_out, int out_size, void* d_ws, size_t ws_size,
                              hipStream_t stream) {
    const float* Xq_f  = (const float*)d_in[0];
    const float* Xk_f  = (const float*)d_in[1];
    const float* Xv_f  = (const float*)d_in[2];
    const unsigned char* mask = (const unsigned char*)d_in[3];
    const float* Wq_f  = (const float*)d_in[4];
    const float* Wk_f  = (const float*)d_in[5];
    const float* Wv_f  = (const float*)d_in[6];
    const float* Wo_f  = (const float*)d_in[7];
    const float* gamma = (const float*)d_in[8];
    const float* beta  = (const float*)d_in[9];

    char* ws = (char*)d_ws;
    size_t off = 0;
    auto take = [&](size_t bytes) -> char* {
        char* p = ws + off;
        off += (bytes + 255) & ~(size_t)255;
        return p;
    };
    const size_t nX  = (size_t)BB * SS * DD;       // 4,194,304
    const size_t nW  = (size_t)DD * HD;            // 131,072
    const size_t nQh = (size_t)BB * HH * SS * DKV; // 8,388,608
    const size_t nP  = (size_t)BB * SS * HD;       // 8,388,608

    bf16_t* Xq  = (bf16_t*)take(nX * 2);
    bf16_t* Xk  = (bf16_t*)take(nX * 2);
    bf16_t* Xv  = (bf16_t*)take(nX * 2);
    bf16_t* WqT = (bf16_t*)take(nW * 2);   // [512][256]
    bf16_t* WkT = (bf16_t*)take(nW * 2);
    bf16_t* WvT = (bf16_t*)take(nW * 2);
    bf16_t* WoT = (bf16_t*)take(nW * 2);   // [256][512]
    bf16_t* Qh  = (bf16_t*)take(nQh * 2);
    bf16_t* Kh  = (bf16_t*)take(nQh * 2);
    bf16_t* VT  = (bf16_t*)take(nQh * 2);
    bf16_t* Pb  = (bf16_t*)take(nP * 2);

    // 0) conversions (+ weight transposes)
    f32_to_bf16_kernel<<<(int)((nX + 255) / 256), 256, 0, stream>>>(Xq_f, Xq, (int)nX);
    f32_to_bf16_kernel<<<(int)((nX + 255) / 256), 256, 0, stream>>>(Xk_f, Xk, (int)nX);
    f32_to_bf16_kernel<<<(int)((nX + 255) / 256), 256, 0, stream>>>(Xv_f, Xv, (int)nX);
    f32_to_bf16_T_kernel<<<(int)((nW + 255) / 256), 256, 0, stream>>>(Wq_f, WqT, DD, HD);
    f32_to_bf16_T_kernel<<<(int)((nW + 255) / 256), 256, 0, stream>>>(Wk_f, WkT, DD, HD);
    f32_to_bf16_T_kernel<<<(int)((nW + 255) / 256), 256, 0, stream>>>(Wv_f, WvT, DD, HD);
    f32_to_bf16_T_kernel<<<(int)((nW + 255) / 256), 256, 0, stream>>>(Wo_f, WoT, HD, DD);

    // 1) projections: 8192 wave strips (16x64) / 8 per block
    proj_gemm_kernel<<<1024, 256, 0, stream>>>(Xq, WqT, Qh, 0);
    proj_gemm_kernel<<<1024, 256, 0, stream>>>(Xk, WkT, Kh, 1);
    proj_gemm_kernel<<<1024, 256, 0, stream>>>(Xv, WvT, VT, 2);

    // 2) attention: B*H*(S/16) = 8192 blocks
    float* outp = (float*)d_out;
    float* attn = outp + nX;                    // second output region
    attn_kernel<<<8192, 128, 0, stream>>>(Qh, Kh, VT, mask, attn, Pb);

    // 3) output projection + residual + layernorm: 16384/16 = 1024 blocks
    out_ln_kernel<<<1024, 128, 0, stream>>>(Pb, WoT, Xq_f, gamma, beta, outp);
}
